// MoE_48679159332904
// MI455X (gfx1250) — matrix operand order repaired and checked by reading, not compiled, not run on hardware
//
#include <hip/hip_runtime.h>
#include <hip/hip_bf16.h>

typedef __attribute__((ext_vector_type(8)))  __bf16 v8bf;
typedef __attribute__((ext_vector_type(16))) __bf16 v16bf;
typedef __attribute__((ext_vector_type(8)))  float  v8f;

#define BB    8
#define TT    2048
#define CC    1024
#define EE    8
#define KTOP  2
#define CAPS  320
#define HH    4096                  // 4*C
#define BT    (BB*TT)               // 16384
#define NSLOT (EE*BB*CAPS)          // 20480

#define LDSTR 72                    // LDS row stride in bf16 elems (144B, 16B-aligned)

// ---------------------------------------------------------------------------
// CDNA5 async global->LDS copy (16B per lane), tracked by ASYNCcnt.
// lds_byte = wave-relative LDS byte offset (low 32 bits of generic pointer).
// ---------------------------------------------------------------------------
__device__ __forceinline__ void async_copy_b128(unsigned lds_byte, const void* gptr) {
    asm volatile("global_load_async_to_lds_b128 %0, %1, off"
                 :: "v"(lds_byte), "v"(gptr) : "memory");
}
__device__ __forceinline__ void wait_async0() {
    asm volatile("s_wait_asynccnt 0x0" ::: "memory");
}
__device__ __forceinline__ unsigned lds_off(const void* p) {
    return (unsigned)(unsigned long long)p;   // LDS aperture: low 32 bits = LDS offset
}

// ---------------------------------------------------------------------------
// fp32 -> bf16 conversion (vectorized by 4) — used for activations x.
// ---------------------------------------------------------------------------
__global__ void moe_cvt_bf16(const float* __restrict__ in, void* __restrict__ outv, long n4) {
    __bf16* out = (__bf16*)outv;
    long i = (long)blockIdx.x * blockDim.x + threadIdx.x;
    long stride = (long)gridDim.x * blockDim.x;
    for (; i < n4; i += stride) {
        float4 f = ((const float4*)in)[i];
        union { __bf16 h[4]; uint2 u; } t;
        t.h[0] = (__bf16)f.x; t.h[1] = (__bf16)f.y;
        t.h[2] = (__bf16)f.z; t.h[3] = (__bf16)f.w;
        ((uint2*)out)[i] = t.u;
    }
}

// ---------------------------------------------------------------------------
// fp32 [R,C] -> bf16 transposed [C,R], LDS-tiled 32x32 (batched over z).
// Removes the 2-byte transpose scatter from the GEMM hot loop.
// ---------------------------------------------------------------------------
__global__ __launch_bounds__(256)
void moe_cvt_transpose(const float* __restrict__ in, void* __restrict__ outv,
                       int R, int C, long sIn, long sOut)
{
    __shared__ float tile[32][33];
    const float* src = in + (long)blockIdx.z * sIn;
    __bf16* out = (__bf16*)outv + (long)blockIdx.z * sOut;
    const int tx = threadIdx.x & 31;
    const int ty = threadIdx.x >> 5;     // 0..7
    const int rb = blockIdx.y * 32, cb = blockIdx.x * 32;
#pragma unroll
    for (int i = 0; i < 4; ++i)
        tile[ty + i * 8][tx] = src[(long)(rb + ty + i * 8) * C + cb + tx];
    __syncthreads();
#pragma unroll
    for (int i = 0; i < 4; ++i)
        out[(long)(cb + ty + i * 8) * R + rb + tx] = (__bf16)tile[tx][ty + i * 8];
}

// ---------------------------------------------------------------------------
// Tiled bf16 WMMA GEMM: C[M,N] = act(A[M,K] @ Bt[N,K]^T + bias), fp32 accum.
// Block = 256 threads (8 waves), tile 128x128, K-step 64 (2x WMMA-K per stage).
// Double-buffered LDS staged via global_load_async_to_lds_b128 (ASYNCcnt),
// overlapping next-stage copies with current-stage v_wmma_f32_16x16x32_bf16.
// blockIdx.z batches over experts via element strides.
// ---------------------------------------------------------------------------
template<bool OUT_BF16, bool RELU>
__global__ __launch_bounds__(256)
void moe_gemm_wmma(const void* __restrict__ Agv, const void* __restrict__ Btv,
                   const float* __restrict__ biasg, void* __restrict__ outv,
                   int M, int N, int Kd,
                   long sA, long sB, long sBias, long sOut)
{
    // per buffer: A tile [128][LDSTR] then B tile [128][LDSTR]
    __shared__ __bf16 sm[2][2 * 128 * LDSTR];

    const int tid  = threadIdx.x;
    const int lane = tid & 31;
    const int wid  = tid >> 5;
    const int mb   = blockIdx.y * 128;
    const int nb   = blockIdx.x * 128;

    const __bf16* A  = (const __bf16*)Agv + (long)blockIdx.z * sA;
    const __bf16* Bt = (const __bf16*)Btv + (long)blockIdx.z * sB;
    const float*  bi = biasg + (long)blockIdx.z * sBias;

    const int wm  = (wid >> 1) * 32;   // wave row offset in tile
    const int wn  = (wid & 1) * 64;    // wave col offset in tile
    const int lh  = lane >> 4;         // lane half (0/1)
    const int l15 = lane & 15;

    v8f acc[2][4];
#pragma unroll
    for (int mi = 0; mi < 2; ++mi)
#pragma unroll
        for (int ni = 0; ni < 4; ++ni)
#pragma unroll
            for (int r = 0; r < 8; ++r) acc[mi][ni][r] = 0.f;

    // cooperative-staging mapping: 2 threads per row, 32 contiguous K each
    const int sr = tid >> 1;            // 0..127
    const int sk = (tid & 1) * 32;      // 0 / 32
    const __bf16* aRow = A  + (long)(mb + sr) * Kd + sk;
    const __bf16* bRow = Bt + (long)(nb + sr) * Kd + sk;

    // ---- async stage of one 64-K slab into LDS buffer `buf` ----
    auto stage = [&](int buf, int ko) {
        unsigned la = lds_off(&sm[buf][sr * LDSTR + sk]);
        unsigned lb = lds_off(&sm[buf][128 * LDSTR + sr * LDSTR + sk]);
        const __bf16* ga = aRow + ko;
        const __bf16* gb = bRow + ko;
#pragma unroll
        for (int c = 0; c < 4; ++c) {
            async_copy_b128(la + c * 16, ga + c * 8);
            async_copy_b128(lb + c * 16, gb + c * 8);
        }
    };

    // ---- consume one staged 64-K slab: 2 x (frag loads + 8 WMMA) ----
    auto compute = [&](int buf) {
        const __bf16* As = &sm[buf][0];
        const __bf16* Bs = &sm[buf][128 * LDSTR];
#pragma unroll
        for (int k32 = 0; k32 < 64; k32 += 32) {
            v16bf af[2], bfv[4];
#pragma unroll
            for (int mi = 0; mi < 2; ++mi) {
                int mr = wm + mi * 16 + l15;
                v8bf lo = *(const v8bf*)&As[mr * LDSTR + k32 + lh * 8];
                v8bf hi = *(const v8bf*)&As[mr * LDSTR + k32 + lh * 8 + 16];
                af[mi] = __builtin_shufflevector(lo, hi, 0,1,2,3,4,5,6,7,8,9,10,11,12,13,14,15);
            }
#pragma unroll
            for (int ni = 0; ni < 4; ++ni) {
                int nc = wn + ni * 16 + l15;
                v8bf lo = *(const v8bf*)&Bs[nc * LDSTR + k32 + lh * 8];
                v8bf hi = *(const v8bf*)&Bs[nc * LDSTR + k32 + lh * 8 + 16];
                bfv[ni] = __builtin_shufflevector(lo, hi, 0,1,2,3,4,5,6,7,8,9,10,11,12,13,14,15);
            }
#pragma unroll
            for (int mi = 0; mi < 2; ++mi)
#pragma unroll
                for (int ni = 0; ni < 4; ++ni)
                    acc[mi][ni] = __builtin_amdgcn_wmma_f32_16x16x32_bf16(
                        false, af[mi], false, bfv[ni],
                        (short)0, acc[mi][ni], false, false);
        }
    };

    // ---- software pipeline: async-prefetch next slab while computing ----
    stage(0, 0);
    wait_async0();
    __syncthreads();
    int buf = 0;
    for (int ko = 0; ko < Kd; ko += 64) {
        const bool more = (ko + 64) < Kd;
        if (more) stage(buf ^ 1, ko + 64);
        compute(buf);
        if (more) wait_async0();
        __syncthreads();
        buf ^= 1;
    }

    // ---- epilogue: bias (+ReLU), bf16 or fp32 store ----
    const long zo = (long)blockIdx.z * sOut;
#pragma unroll
    for (int mi = 0; mi < 2; ++mi) {
#pragma unroll
        for (int ni = 0; ni < 4; ++ni) {
            int col = nb + wn + ni * 16 + l15;
            float bv = bi[col];
#pragma unroll
            for (int r = 0; r < 8; ++r) {
                int row = mb + wm + mi * 16 + lh * 8 + r;
                float v = acc[mi][ni][r] + bv;
                if (RELU) v = v > 0.f ? v : 0.f;
                long idx = zo + (long)row * N + col;
                if (OUT_BF16) ((__bf16*)outv)[idx] = (__bf16)v;
                else          ((float*)outv)[idx]  = v;
            }
        }
    }
}

// ---------------------------------------------------------------------------
// Router tail: logits = H2 @ Wr3 + br3 ; softmax ; top-2 (first-occurrence ties)
// One wave per token; 8 waves per block.
// ---------------------------------------------------------------------------
__global__ __launch_bounds__(256)
void moe_router_topk(const void* __restrict__ H2v, const float* __restrict__ Wr3,
                     const float* __restrict__ br3,
                     int* __restrict__ topkE, float* __restrict__ topkP)
{
    const __bf16* H2 = (const __bf16*)H2v;
    const int lane  = threadIdx.x & 31;
    const int wid   = threadIdx.x >> 5;
    const int token = blockIdx.x * 8 + wid;

    float acc[EE];
#pragma unroll
    for (int e = 0; e < EE; ++e) acc[e] = 0.f;

    const __bf16* hrow = H2 + (long)token * HH;
    for (int k = lane; k < HH; k += 32) {
        float xv = (float)hrow[k];
        const float4* w = (const float4*)&Wr3[(long)k * EE];
        float4 w0 = w[0], w1 = w[1];
        acc[0] += xv * w0.x; acc[1] += xv * w0.y;
        acc[2] += xv * w0.z; acc[3] += xv * w0.w;
        acc[4] += xv * w1.x; acc[5] += xv * w1.y;
        acc[6] += xv * w1.z; acc[7] += xv * w1.w;
    }
#pragma unroll
    for (int off = 16; off > 0; off >>= 1)
#pragma unroll
        for (int e = 0; e < EE; ++e)
            acc[e] += __shfl_down(acc[e], off, 32);

    if (lane == 0) {
        float lg[EE], p[EE];
        float mx = -1e30f;
#pragma unroll
        for (int e = 0; e < EE; ++e) { lg[e] = acc[e] + br3[e]; mx = lg[e] > mx ? lg[e] : mx; }
        float s = 0.f;
#pragma unroll
        for (int e = 0; e < EE; ++e) { p[e] = __expf(lg[e] - mx); s += p[e]; }
        float inv = 1.f / s;
#pragma unroll
        for (int e = 0; e < EE; ++e) p[e] *= inv;
        int e0 = 0;
#pragma unroll
        for (int e = 1; e < EE; ++e) if (p[e] > p[e0]) e0 = e;
        int e1 = (e0 == 0) ? 1 : 0;
#pragma unroll
        for (int e = 0; e < EE; ++e) if (e != e0 && e != e1 && p[e] > p[e1]) e1 = e;
        topkE[token * 2 + 0] = e0;  topkE[token * 2 + 1] = e1;
        topkP[token * 2 + 0] = p[e0]; topkP[token * 2 + 1] = p[e1];
    }
}

// ---------------------------------------------------------------------------
// Per-batch-row dispatch scan: row-major cumsum over (t,k) via wave ballots.
// One wave (32 threads) per batch row b.
// ---------------------------------------------------------------------------
__global__ __launch_bounds__(32)
void moe_scan_dispatch(const int* __restrict__ topkE,
                       int* __restrict__ dest, int* __restrict__ slotSrc)
{
    const int b = blockIdx.x;
    const int lane = threadIdx.x;

    for (int i = lane; i < EE * CAPS; i += 32) {
        int e = i / CAPS, pp = i % CAPS;
        slotSrc[e * (BB * CAPS) + b * CAPS + pp] = -1;
    }
    __syncthreads();

    int cnt[EE];
#pragma unroll
    for (int e = 0; e < EE; ++e) cnt[e] = 0;

    const int iters = (TT * KTOP) / 32;   // 128
    for (int it = 0; it < iters; ++it) {
        int idx = it * 32 + lane;                     // flat (t,k)
        int e   = topkE[b * (TT * KTOP) + idx];
        unsigned long long bal[EE];
#pragma unroll
        for (int j = 0; j < EE; ++j) bal[j] = __ballot(e == j);
        unsigned long long below = ((unsigned long long)1 << lane) - 1ull;
        int pos = cnt[e] + (int)__popcll(bal[e] & below);
        bool keep = pos < CAPS;
        int d = keep ? (e * (BB * CAPS) + b * CAPS + pos) : -1;
        dest[b * (TT * KTOP) + idx] = d;
        if (keep) slotSrc[d] = b * TT + (idx >> 1);   // source token row
#pragma unroll
        for (int j = 0; j < EE; ++j) cnt[j] += (int)__popcll(bal[j]);
    }
}

// ---------------------------------------------------------------------------
// Gather tokens into padded per-(expert,row) buffers (bf16), zeros for padding.
// ---------------------------------------------------------------------------
__global__ __launch_bounds__(256)
void moe_gather(const void* __restrict__ Xbfv, const int* __restrict__ slotSrc,
                void* __restrict__ Xinv)
{
    const __bf16* Xbf = (const __bf16*)Xbfv;
    __bf16* Xin = (__bf16*)Xinv;
    const int slot = blockIdx.x;
    const int src  = slotSrc[slot];
    const int c4   = threadIdx.x * 4;
    uint2 v = make_uint2(0u, 0u);
    if (src >= 0) v = *(const uint2*)&Xbf[(long)src * CC + c4];
    *(uint2*)&Xin[(long)slot * CC + c4] = v;
}

// ---------------------------------------------------------------------------
// Combine: out[token,c] = sum_k p_k * EY[dest_k, c]  (dropped slots contribute 0)
// ---------------------------------------------------------------------------
__global__ __launch_bounds__(256)
void moe_combine(const float* __restrict__ EY, const int* __restrict__ dest,
                 const float* __restrict__ topkP, float* __restrict__ out)
{
    const int token = blockIdx.x;
    const int c = threadIdx.x * 4;
    int   d0 = dest[token * 2 + 0], d1 = dest[token * 2 + 1];
    float p0 = topkP[token * 2 + 0], p1 = topkP[token * 2 + 1];
    float4 v = make_float4(0.f, 0.f, 0.f, 0.f);
    if (d0 >= 0) {
        float4 a = *(const float4*)&EY[(long)d0 * CC + c];
        v.x += p0 * a.x; v.y += p0 * a.y; v.z += p0 * a.z; v.w += p0 * a.w;
    }
    if (d1 >= 0) {
        float4 a = *(const float4*)&EY[(long)d1 * CC + c];
        v.x += p1 * a.x; v.y += p1 * a.y; v.z += p1 * a.z; v.w += p1 * a.w;
    }
    *(float4*)&out[(long)token * CC + c] = v;
}

// ---------------------------------------------------------------------------
// Host launch
// ---------------------------------------------------------------------------
extern "C" void kernel_launch(void* const* d_in, const int* in_sizes, int n_in,
                              void* d_out, int out_size, void* d_ws, size_t ws_size,
                              hipStream_t stream) {
    const float* x   = (const float*)d_in[0];
    const float* Wr1 = (const float*)d_in[1];
    const float* br1 = (const float*)d_in[2];
    const float* Wr2 = (const float*)d_in[3];
    const float* br2 = (const float*)d_in[4];
    const float* Wr3 = (const float*)d_in[5];
    const float* br3 = (const float*)d_in[6];
    const float* We1 = (const float*)d_in[7];
    const float* be1 = (const float*)d_in[8];
    const float* We2 = (const float*)d_in[9];
    const float* be2 = (const float*)d_in[10];
    float* out = (float*)d_out;

    unsigned char* w = (unsigned char*)d_ws;
    size_t off = 0;
    auto take = [&](size_t bytes) -> void* {
        void* p = w + off;
        off = (off + bytes + 255) & ~(size_t)255;
        return p;
    };
    void* Xbf    = take((size_t)BT * CC * 2);           // x as bf16 [BT,C]
    void* Wr1t   = take((size_t)CC * HH * 2);           // bf16 [HH,CC]
    void* Wr2t   = take((size_t)HH * HH * 2);           // bf16 [HH,HH]
    void* We1t   = take((size_t)EE * CC * HH * 2);      // bf16 [E][HH,CC]
    void* We2t   = take((size_t)EE * HH * CC * 2);      // bf16 [E][CC,HH]
    void* H1bf   = take((size_t)BT * HH * 2);
    void* H2bf   = take((size_t)BT * HH * 2);
    int*   topkE   = (int*)  take((size_t)BT * KTOP * 4);
    float* topkP   = (float*)take((size_t)BT * KTOP * 4);
    int*   dest    = (int*)  take((size_t)BT * KTOP * 4);
    int*   slotSrc = (int*)  take((size_t)NSLOT * 4);
    void* Xin    = take((size_t)NSLOT * CC * 2);
    void* EHbf   = take((size_t)NSLOT * HH * 2);
    float* EY    = (float*)take((size_t)NSLOT * CC * 4);

    // ---- 1. convert x to bf16; convert+transpose all GEMM weights ----
    moe_cvt_bf16<<<dim3(2048), dim3(256), 0, stream>>>(x, Xbf, (long)BT * CC / 4);
    moe_cvt_transpose<<<dim3(HH / 32, CC / 32, 1), dim3(256), 0, stream>>>(
        Wr1, Wr1t, CC, HH, 0, 0);
    moe_cvt_transpose<<<dim3(HH / 32, HH / 32, 1), dim3(256), 0, stream>>>(
        Wr2, Wr2t, HH, HH, 0, 0);
    moe_cvt_transpose<<<dim3(HH / 32, CC / 32, EE), dim3(256), 0, stream>>>(
        We1, We1t, CC, HH, (long)CC * HH, (long)CC * HH);
    moe_cvt_transpose<<<dim3(CC / 32, HH / 32, EE), dim3(256), 0, stream>>>(
        We2, We2t, HH, CC, (long)HH * CC, (long)HH * CC);

    // ---- 2. router GEMM1: H1 = relu(x @ Wr1 + br1)   [16384 x 4096] ----
    moe_gemm_wmma<true, true><<<dim3(HH / 128, BT / 128, 1), dim3(256), 0, stream>>>(
        Xbf, Wr1t, br1, H1bf, BT, HH, CC, 0, 0, 0, 0);

    // ---- 3. router GEMM2: H2 = relu(H1 @ Wr2 + br2)  [16384 x 4096] ----
    moe_gemm_wmma<true, true><<<dim3(HH / 128, BT / 128, 1), dim3(256), 0, stream>>>(
        H1bf, Wr2t, br2, H2bf, BT, HH, HH, 0, 0, 0, 0);

    // ---- 4. router logits + softmax + top-2 ----
    moe_router_topk<<<dim3(BT / 8), dim3(256), 0, stream>>>(H2bf, Wr3, br3, topkE, topkP);

    // ---- 5. capacity scan / dispatch ----
    moe_scan_dispatch<<<dim3(BB), dim3(32), 0, stream>>>(topkE, dest, slotSrc);

    // ---- 6. gather into padded expert buffers ----
    moe_gather<<<dim3(NSLOT), dim3(256), 0, stream>>>(Xbf, slotSrc, Xin);

    // ---- 7. expert GEMM1: EH = relu(Xin @ We1[e] + be1[e])  [8 x 2560 x 4096] ----
    moe_gemm_wmma<true, true><<<dim3(HH / 128, (BB * CAPS) / 128, EE), dim3(256), 0, stream>>>(
        Xin, We1t, be1, EHbf, BB * CAPS, HH, CC,
        (long)(BB * CAPS) * CC, (long)CC * HH, (long)HH, (long)(BB * CAPS) * HH);

    // ---- 8. expert GEMM2: EY = EH @ We2[e] + be2[e]  [8 x 2560 x 1024] fp32 out ----
    moe_gemm_wmma<false, false><<<dim3(CC / 128, (BB * CAPS) / 128, EE), dim3(256), 0, stream>>>(
        EHbf, We2t, be2, EY, BB * CAPS, CC, HH,
        (long)(BB * CAPS) * HH, (long)HH * CC, (long)CC, (long)(BB * CAPS) * CC);

    // ---- 9. weighted combine ----
    moe_combine<<<dim3(BT), dim3(256), 0, stream>>>(EY, dest, topkP, out);
}
